// Decoder_3753801416876
// MI455X (gfx1250) — matrix operand order repaired and checked by reading, not compiled, not run on hardware
//
#include <hip/hip_runtime.h>
#include <hip/hip_bf16.h>

// Problem dims (fixed by the reference)
#define E_DIM 512
#define H_DIM 512
#define B_DIM 64
#define S_DIM 1024
#define T_DIM 256

typedef __bf16 bf16;
typedef __attribute__((ext_vector_type(16))) __bf16 v16bf;
typedef __attribute__((ext_vector_type(8)))  float  v8f;

union ABPack  { uint4 u[2]; v16bf v; };
union BF4Pack { uint2 u; bf16 h[4]; };
union F16Pack { float4 f4[4]; float f[16]; };
union PK16    { uint4 u[2]; bf16 h[16]; };

// ---------------------------------------------------------------------------
// Paired-N WMMA GEMM core: out[m, n] = sum_k A[m,k] * W[n,k]  (A @ W^T)
// One wave computes a 16x32 output tile (two 16x16 accumulators sharing the
// A fetch -> 2 WMMAs per A load, half the A traffic of the 1-tile version).
// A: (M x K) bf16 row-major; W: (N x K) bf16 row-major; K % 32 == 0; N % 32 == 0.
// ---------------------------------------------------------------------------
template <typename OutT>
__device__ __forceinline__ void gemm_pair_body(
    int tilepair, int lane,
    const bf16* __restrict__ A, int lda,
    const bf16* __restrict__ W, int ldw,
    OutT* __restrict__ out, long ldo, long out_base,
    int NtilePairs, int K)
{
    int np = tilepair % NtilePairs;
    int mt = tilepair / NtilePairs;
    const int  l  = lane & 15;
    const bool hi = lane >= 16;

    // A lane layout (16x32 bf16): row = mt*16 + l;
    //   elems 0..15 -> K = k0 + (hi?16:0) + e     (32B contiguous)
    const bf16* arow  = A + (size_t)(mt * 16 + l) * (size_t)lda + (hi ? 16 : 0);
    // B lane layout (32x16 bf16): col = ntile*16 + l; elems -> K = k0 + (hi?16:0) + e
    const bf16* wrow0 = W + (size_t)(np * 32 + l) * (size_t)ldw + (hi ? 16 : 0);
    const bf16* wrow1 = wrow0 + (size_t)16 * (size_t)ldw;

    v8f acc0 = {}, acc1 = {};
    for (int k = 0; k < K; k += 32) {
        ABPack a, b0, b1;
        a.u[0]  = *(const uint4*)(arow  + k);
        a.u[1]  = *(const uint4*)(arow  + k + 8);
        b0.u[0] = *(const uint4*)(wrow0 + k);
        b0.u[1] = *(const uint4*)(wrow0 + k + 8);
        b1.u[0] = *(const uint4*)(wrow1 + k);
        b1.u[1] = *(const uint4*)(wrow1 + k + 8);
        acc0 = __builtin_amdgcn_wmma_f32_16x16x32_bf16(
                   false, a.v, false, b0.v, (short)0, acc0, false, false);
        acc1 = __builtin_amdgcn_wmma_f32_16x16x32_bf16(
                   false, a.v, false, b1.v, (short)0, acc1, false, false);
    }

    // C/D layout: VGPR i -> M = i (+8 for hi lanes), N = lane%16
    long col0 = (long)(np * 32 + l);
#pragma unroll
    for (int i = 0; i < 8; ++i) {
        long row = (long)(mt * 16) + i + (hi ? 8 : 0);
        out[out_base + row * ldo + col0]      = (OutT)acc0[i];
        out[out_base + row * ldo + col0 + 16] = (OutT)acc1[i];
    }
}

template <typename OutT>
__global__ void wmma_gemm_pair(const bf16* __restrict__ A, int lda,
                               const bf16* __restrict__ W, int ldw,
                               OutT* __restrict__ out, long ldo, long out_base,
                               int totalTilePairs, int NtilePairs, int K)
{
    int tp = (int)((blockIdx.x * blockDim.x + threadIdx.x) >> 5);
    if (tp >= totalTilePairs) return;     // wave-uniform: EXEC all-ones for WMMA
    gemm_pair_body<OutT>(tp, threadIdx.x & 31, A, lda, W, ldw,
                         out, ldo, out_base, NtilePairs, K);
}

// ---------------------------------------------------------------------------
// Fused GRU GEMMs in one launch:
//   gi = xcat @ W_ih^T   (64 x 1536, K=1536)  -> tile-pairs [0, 192)
//   gh = h    @ W_hh^T   (64 x 1536, K= 512)  -> tile-pairs [192, 384)
// grid: 48 blocks x 256 threads = 384 waves
// ---------------------------------------------------------------------------
__global__ void gru_gemms(const bf16* __restrict__ xcat, const bf16* __restrict__ wih,
                          const bf16* __restrict__ h_bf, const bf16* __restrict__ whh,
                          float* __restrict__ gi, float* __restrict__ gh)
{
    int tp   = (int)((blockIdx.x * blockDim.x + threadIdx.x) >> 5);
    int lane = threadIdx.x & 31;
    if (tp < 192)
        gemm_pair_body<float>(tp, lane, xcat, E_DIM + 2 * H_DIM, wih, E_DIM + 2 * H_DIM,
                              gi, (long)(3 * H_DIM), 0L, 48, E_DIM + 2 * H_DIM);
    else
        gemm_pair_body<float>(tp - 192, lane, h_bf, H_DIM, whh, H_DIM,
                              gh, (long)(3 * H_DIM), 0L, 48, H_DIM);
}

// ---------------------------------------------------------------------------
// f32 -> bf16 conversion (grid-stride)
// ---------------------------------------------------------------------------
__global__ void cvt_f32_bf16(const float* __restrict__ in, bf16* __restrict__ out, long n)
{
    long i = (long)blockIdx.x * blockDim.x + threadIdx.x;
    long stride = (long)gridDim.x * blockDim.x;
    for (; i < n; i += stride) out[i] = (bf16)in[i];
}

// ---------------------------------------------------------------------------
// hidden0 = tanh(bridge_raw + b_bridge)  -> h_f32, h_bf16      (64*512 elems)
// ---------------------------------------------------------------------------
__global__ void bridge_act(const float* __restrict__ raw, const float* __restrict__ b_bridge,
                           float* __restrict__ h_f32, bf16* __restrict__ h_bf16)
{
    int i = blockIdx.x * blockDim.x + threadIdx.x;
    int j = i & (H_DIM - 1);
    float h = tanhf(raw[i] + b_bridge[j]);
    h_f32[i]  = h;
    h_bf16[i] = (bf16)h;
}

// ---------------------------------------------------------------------------
// scores[b,s] = sum_h v[h] * tanh(q[b,h] + proj_key[b,s,h]), masked.
// Lane owns a contiguous 16-wide h-chunk (2 x b128 per row); q and v_energy
// are hoisted into registers and reused across 4 rows per wave.
// grid (32, B), block 256: 8 waves x 4 rows = 32 rows per block.
// ---------------------------------------------------------------------------
__global__ void attn_scores(const float* __restrict__ q, const bf16* __restrict__ pk,
                            const float* __restrict__ v_energy,
                            const unsigned char* __restrict__ mask,
                            float* __restrict__ scores)
{
    int lane = threadIdx.x & 31;
    int wave = threadIdx.x >> 5;
    int b  = blockIdx.y;
    int s0 = (blockIdx.x * 8 + wave) * 4;

    F16Pack qv, vv;
    const float4* qp = (const float4*)(q + (size_t)b * H_DIM + lane * 16);
    const float4* vp = (const float4*)(v_energy + lane * 16);
#pragma unroll
    for (int j = 0; j < 4; ++j) { qv.f4[j] = qp[j]; vv.f4[j] = vp[j]; }

    for (int r = 0; r < 4; ++r) {
        int s = s0 + r;
        const bf16* p = pk + ((size_t)b * S_DIM + s) * H_DIM + lane * 16;
        PK16 pv;
        pv.u[0] = *(const uint4*)(p);
        pv.u[1] = *(const uint4*)(p + 8);
        float acc = 0.f;
#pragma unroll
        for (int e = 0; e < 16; ++e)
            acc += vv.f[e] * tanhf(qv.f[e] + (float)pv.h[e]);
        for (int off = 16; off; off >>= 1) acc += __shfl_xor(acc, off, 32);
        if (lane == 0)
            scores[(size_t)b * S_DIM + s] =
                mask[(size_t)b * S_DIM + s] ? acc : -3.402823466e38f;
    }
}

// ---------------------------------------------------------------------------
// One block per batch b: softmax over S=1024, context[b,:] = alpha @ eh[b],
// plus the per-step embed copy (from pre-converted bf16 trg_embed).
// Thread owns 4 contiguous d columns -> one b64 load per s-iteration.
// ---------------------------------------------------------------------------
__global__ void softmax_context(const float* __restrict__ scores, const bf16* __restrict__ eh,
                                const bf16* __restrict__ trg_bf, int t,
                                bf16* __restrict__ xcat, bf16* __restrict__ pcat)
{
    __shared__ float al[S_DIM];
    __shared__ float red[8];
    int b = blockIdx.x;
    int tid = threadIdx.x;                 // 256 threads

    // ---- fused embed copy: xcat[:,0:512], pcat[:,0:512] (2 bf16 per thread) ----
    {
        const bf16* erow = trg_bf + ((size_t)b * T_DIM + t) * E_DIM;
        unsigned int ev = *(const unsigned int*)(erow + tid * 2);
        *(unsigned int*)(xcat + (size_t)b * (E_DIM + 2 * H_DIM) + tid * 2) = ev;
        *(unsigned int*)(pcat + (size_t)b * (E_DIM + 3 * H_DIM) + tid * 2) = ev;
    }

    // ---- softmax ----
    float sv[4];
    float m = -3.402823466e38f;
#pragma unroll
    for (int j = 0; j < 4; ++j) {
        sv[j] = scores[(size_t)b * S_DIM + tid + j * 256];
        m = fmaxf(m, sv[j]);
    }
    for (int off = 16; off; off >>= 1) m = fmaxf(m, __shfl_xor(m, off, 32));
    if ((tid & 31) == 0) red[tid >> 5] = m;
    __syncthreads();
    float bm = red[0];
#pragma unroll
    for (int w = 1; w < 8; ++w) bm = fmaxf(bm, red[w]);

    float ssum = 0.f;
#pragma unroll
    for (int j = 0; j < 4; ++j) {
        float e = __expf(sv[j] - bm);
        al[tid + j * 256] = e;
        ssum += e;
    }
    for (int off = 16; off; off >>= 1) ssum += __shfl_xor(ssum, off, 32);
    __syncthreads();                       // done reading red[] (max phase)
    if ((tid & 31) == 0) red[tid >> 5] = ssum;
    __syncthreads();
    float bs = 0.f;
#pragma unroll
    for (int w = 0; w < 8; ++w) bs += red[w];
    float inv = 1.0f / bs;
#pragma unroll
    for (int j = 0; j < 4; ++j) al[tid + j * 256] *= inv;
    __syncthreads();

    // ---- context: thread owns d = tid*4 .. tid*4+3 (contiguous, b64 loads) ----
    float c[4] = {0.f, 0.f, 0.f, 0.f};
    const bf16* ehb = eh + (size_t)b * S_DIM * (2 * H_DIM) + tid * 4;
    for (int s2 = 0; s2 < S_DIM; ++s2) {
        float a = al[s2];
        BF4Pack r;
        r.u = *(const uint2*)(ehb + (size_t)s2 * (2 * H_DIM));
        c[0] += a * (float)r.h[0];
        c[1] += a * (float)r.h[1];
        c[2] += a * (float)r.h[2];
        c[3] += a * (float)r.h[3];
    }
    BF4Pack cw;
#pragma unroll
    for (int j = 0; j < 4; ++j) cw.h[j] = (bf16)c[j];
    *(uint2*)(xcat + (size_t)b * (E_DIM + 2 * H_DIM) + E_DIM + tid * 4)         = cw.u;
    *(uint2*)(pcat + (size_t)b * (E_DIM + 3 * H_DIM) + E_DIM + H_DIM + tid * 4) = cw.u;
}

// ---------------------------------------------------------------------------
// GRU gate fusion; writes h (f32 + bf16), decoder_states[:,t,:], pcat[:,512:1024]
// ---------------------------------------------------------------------------
__global__ void gru_gates(const float* __restrict__ gi, const float* __restrict__ gh,
                          const float* __restrict__ b_ih, const float* __restrict__ b_hh,
                          float* __restrict__ h_f32, bf16* __restrict__ h_bf16,
                          bf16* __restrict__ pcat, float* __restrict__ dec_states, int t)
{
    int i = blockIdx.x * blockDim.x + threadIdx.x;   // 0 .. 64*512-1
    int b = i >> 9, j = i & 511;
    const float* gib = gi + (size_t)b * (3 * H_DIM);
    const float* ghb = gh + (size_t)b * (3 * H_DIM);
    float ir  = gib[j]             + b_ih[j];
    float iz  = gib[j + H_DIM]     + b_ih[j + H_DIM];
    float in_ = gib[j + 2 * H_DIM] + b_ih[j + 2 * H_DIM];
    float hr  = ghb[j]             + b_hh[j];
    float hz  = ghb[j + H_DIM]     + b_hh[j + H_DIM];
    float hn  = ghb[j + 2 * H_DIM] + b_hh[j + 2 * H_DIM];
    float r = 1.f / (1.f + __expf(-(ir + hr)));
    float z = 1.f / (1.f + __expf(-(iz + hz)));
    float n = tanhf(in_ + r * hn);
    float hnew = (1.f - z) * n + z * h_f32[i];
    h_f32[i]  = hnew;
    h_bf16[i] = (bf16)hnew;
    pcat[(size_t)b * (E_DIM + 3 * H_DIM) + E_DIM + j] = (bf16)hnew;
    dec_states[((size_t)b * T_DIM + t) * H_DIM + j] = hnew;
}

__global__ void copy_hidden_final(const float* __restrict__ h, float* __restrict__ out)
{
    int i = blockIdx.x * blockDim.x + threadIdx.x;   // 64*512
    out[i] = h[i];
}

// ---------------------------------------------------------------------------
// Workspace layout (bytes; every size is a multiple of 256)
// ---------------------------------------------------------------------------
#define WS_EH     ((size_t)0)                                   // 64*1024*1024*2 = 134217728
#define WS_PK     (WS_EH    + (size_t)134217728)                // 64*1024* 512*2 =  67108864
#define WS_WKEY   (WS_PK    + (size_t)67108864)                 // 512*1024*2 = 1048576
#define WS_WQ     (WS_WKEY  + (size_t)1048576)                  // 512* 512*2 =  524288
#define WS_WIH    (WS_WQ    + (size_t)524288)                   // 1536*1536*2 = 4718592
#define WS_WHH    (WS_WIH   + (size_t)4718592)                  // 1536* 512*2 = 1572864
#define WS_WPRE   (WS_WHH   + (size_t)1572864)                  // 512*2048*2 = 2097152
#define WS_WBR    (WS_WPRE  + (size_t)2097152)                  // 512*1024*2 = 1048576
#define WS_EF     (WS_WBR   + (size_t)1048576)                  // 64*1024*2 =  131072
#define WS_HF32   (WS_EF    + (size_t)131072)                   // 64*512*4  =  131072
#define WS_HBF16  (WS_HF32  + (size_t)131072)                   // 64*512*2  =   65536
#define WS_Q      (WS_HBF16 + (size_t)65536)                    // 64*512*4  =  131072
#define WS_SCORES (WS_Q     + (size_t)131072)                   // 64*1024*4 =  262144
#define WS_XCAT   (WS_SCORES+ (size_t)262144)                   // 64*1536*2 =  196608
#define WS_PCAT   (WS_XCAT  + (size_t)196608)                   // 64*2048*2 =  262144
#define WS_GI     (WS_PCAT  + (size_t)262144)                   // 64*1536*4 =  393216
#define WS_GH     (WS_GI    + (size_t)393216)                   // 64*1536*4 =  393216
#define WS_BRAW   (WS_GH    + (size_t)393216)                   // 64*512*4  =  131072
#define WS_TRG    (WS_BRAW  + (size_t)131072)                   // 64*256*512*2 = 16777216

static inline int cvt_blocks(long n)
{
    long b = (n + 255) / 256;
    return (int)(b > 8192 ? 8192 : b);
}

extern "C" void kernel_launch(void* const* d_in, const int* in_sizes, int n_in,
                              void* d_out, int out_size, void* d_ws, size_t ws_size,
                              hipStream_t stream)
{
    (void)in_sizes; (void)n_in; (void)out_size; (void)ws_size;

    const float* trg_embed      = (const float*)d_in[0];
    const float* encoder_hidden = (const float*)d_in[1];
    const float* encoder_final  = (const float*)d_in[2];
    const unsigned char* src_mask = (const unsigned char*)d_in[3];
    /* d_in[4] trg_mask unused by reference */
    const float* W_key    = (const float*)d_in[5];
    const float* W_query  = (const float*)d_in[6];
    const float* v_energy = (const float*)d_in[7];
    const float* W_bridge = (const float*)d_in[8];
    const float* b_bridge = (const float*)d_in[9];
    const float* W_ih     = (const float*)d_in[10];
    const float* W_hh     = (const float*)d_in[11];
    const float* b_ih     = (const float*)d_in[12];
    const float* b_hh     = (const float*)d_in[13];
    const float* W_pre    = (const float*)d_in[14];

    char* ws = (char*)d_ws;
    bf16*  eh_bf   = (bf16*)(ws + WS_EH);
    bf16*  pk_bf   = (bf16*)(ws + WS_PK);
    bf16*  wkey_bf = (bf16*)(ws + WS_WKEY);
    bf16*  wq_bf   = (bf16*)(ws + WS_WQ);
    bf16*  wih_bf  = (bf16*)(ws + WS_WIH);
    bf16*  whh_bf  = (bf16*)(ws + WS_WHH);
    bf16*  wpre_bf = (bf16*)(ws + WS_WPRE);
    bf16*  wbr_bf  = (bf16*)(ws + WS_WBR);
    bf16*  ef_bf   = (bf16*)(ws + WS_EF);
    float* h_f32   = (float*)(ws + WS_HF32);
    bf16*  h_bf    = (bf16*)(ws + WS_HBF16);
    float* q_f32   = (float*)(ws + WS_Q);
    float* scores  = (float*)(ws + WS_SCORES);
    bf16*  xcat    = (bf16*)(ws + WS_XCAT);
    bf16*  pcat    = (bf16*)(ws + WS_PCAT);
    float* gi      = (float*)(ws + WS_GI);
    float* gh      = (float*)(ws + WS_GH);
    float* braw    = (float*)(ws + WS_BRAW);
    bf16*  trg_bf  = (bf16*)(ws + WS_TRG);

    // d_out layout: decoder_states (B,T,H) | hidden_final (1,B,H) | pre_output_vectors (B,T,H)
    float* dec_states = (float*)d_out;
    float* hid_final  = (float*)d_out + (size_t)B_DIM * T_DIM * H_DIM;
    float* pre_out    = hid_final + (size_t)B_DIM * H_DIM;

    const dim3 blk(256);

    // ---- one-time: f32 -> bf16 conversions ----
    long n_eh = (long)B_DIM * S_DIM * 2 * H_DIM;
    cvt_f32_bf16<<<cvt_blocks(n_eh), blk, 0, stream>>>(encoder_hidden, eh_bf, n_eh);
    long n_trg = (long)B_DIM * T_DIM * E_DIM;
    cvt_f32_bf16<<<cvt_blocks(n_trg), blk, 0, stream>>>(trg_embed, trg_bf, n_trg);
    cvt_f32_bf16<<<cvt_blocks(64*1024),  blk, 0, stream>>>(encoder_final, ef_bf, 64*1024);
    cvt_f32_bf16<<<cvt_blocks(512*1024), blk, 0, stream>>>(W_key,    wkey_bf, 512*1024);
    cvt_f32_bf16<<<cvt_blocks(512*512),  blk, 0, stream>>>(W_query,  wq_bf,   512*512);
    cvt_f32_bf16<<<cvt_blocks(1536*1536),blk, 0, stream>>>(W_ih,     wih_bf,  1536*1536);
    cvt_f32_bf16<<<cvt_blocks(1536*512), blk, 0, stream>>>(W_hh,     whh_bf,  1536*512);
    cvt_f32_bf16<<<cvt_blocks(512*2048), blk, 0, stream>>>(W_pre,    wpre_bf, 512*2048);
    cvt_f32_bf16<<<cvt_blocks(512*1024), blk, 0, stream>>>(W_bridge, wbr_bf,  512*1024);

    // ---- one-time: proj_key = encoder_hidden @ W_key^T (65536 x 512, K=1024), bf16 out
    {
        int tp = 4096 * 16;   // Mtiles(4096) x NtilePairs(16)
        wmma_gemm_pair<bf16><<<(tp + 7) / 8, blk, 0, stream>>>(
            eh_bf, 2 * H_DIM, wkey_bf, 2 * H_DIM, pk_bf, (long)H_DIM, 0L, tp, 16, 2 * H_DIM);
    }

    // ---- one-time: hidden0 = tanh(encoder_final @ W_bridge^T + b_bridge)
    {
        int tp = 4 * 16;
        wmma_gemm_pair<float><<<(tp + 7) / 8, blk, 0, stream>>>(
            ef_bf, 2 * H_DIM, wbr_bf, 2 * H_DIM, braw, (long)H_DIM, 0L, tp, 16, 2 * H_DIM);
        bridge_act<<<(B_DIM * H_DIM) / 256, blk, 0, stream>>>(braw, b_bridge, h_f32, h_bf);
    }

    // ---- sequential scan over T steps (6 launches per step) ----
    for (int t = 0; t < T_DIM; ++t) {
        // q = h @ W_query^T  (64 x 512, K=512)
        wmma_gemm_pair<float><<<8, blk, 0, stream>>>(
            h_bf, H_DIM, wq_bf, H_DIM, q_f32, (long)H_DIM, 0L, 64, 16, H_DIM);

        attn_scores<<<dim3(32, B_DIM), blk, 0, stream>>>(
            q_f32, pk_bf, v_energy, src_mask, scores);

        softmax_context<<<B_DIM, blk, 0, stream>>>(scores, eh_bf, trg_bf, t, xcat, pcat);

        gru_gemms<<<48, blk, 0, stream>>>(xcat, wih_bf, h_bf, whh_bf, gi, gh);

        gru_gates<<<(B_DIM * H_DIM) / 256, blk, 0, stream>>>(
            gi, gh, b_ih, b_hh, h_f32, h_bf, pcat, dec_states, t);

        // pre = [embed, h_new, ctx] @ W_pre^T (64 x 512, K=2048) straight into d_out
        wmma_gemm_pair<float><<<8, blk, 0, stream>>>(
            pcat, E_DIM + 3 * H_DIM, wpre_bf, E_DIM + 3 * H_DIM,
            pre_out, (long)(T_DIM * H_DIM), (long)t * H_DIM, 64, 16, E_DIM + 3 * H_DIM);
    }

    copy_hidden_final<<<(B_DIM * H_DIM) / 256, blk, 0, stream>>>(h_f32, hid_final);
}
